// TransformerBlock_38448547234133
// MI455X (gfx1250) — matrix operand order repaired and checked by reading, not compiled, not run on hardware
//
#include <hip/hip_runtime.h>

// ---------------------------------------------------------------------------
// MI455X (gfx1250) transformer block: WMMA f16 GEMMs + implicit-GEMM convs.
// Wave32, 16x16x32 f16 WMMA, f32 accumulate.
// Block tile 128x128, 8 waves (4x2), wave tile 32x64 (2x4 WMMA frags).
// LDS staging via GLOBAL_LOAD_ASYNC_TO_LDS_B128 (ASYNCcnt-tracked).
// ---------------------------------------------------------------------------

typedef __attribute__((ext_vector_type(16))) _Float16 v16h;
typedef __attribute__((ext_vector_type(8)))  _Float16 v8h;
typedef __attribute__((ext_vector_type(8)))  float    v8f;
typedef int v4i __attribute__((vector_size(16)));

union HFrag { v16h v; v8h h[2]; };

constexpr int Bn  = 8;     // batch
constexpr int Cc  = 256;   // channels
constexpr int Nn  = 4096;  // tokens (16^3)
constexpr int NHh = 8;     // heads
constexpr int DHh = 32;    // head dim
constexpr float EPSv = 1e-5f;

#if defined(__has_builtin)
#  if __has_builtin(__builtin_amdgcn_global_load_async_to_lds_b128)
#    define USE_ASYNC_LDS 1
#  endif
#endif
#ifndef USE_ASYNC_LDS
#  define USE_ASYNC_LDS 0
#endif

// Stage 16 halves (32B) global -> LDS.
__device__ __forceinline__ void stage16(const _Float16* __restrict__ g, _Float16* l) {
#if USE_ASYNC_LDS
  __attribute__((address_space(1))) v4i* gp =
      (__attribute__((address_space(1))) v4i*)(const_cast<_Float16*>(g));
  __attribute__((address_space(3))) v4i* lp =
      (__attribute__((address_space(3))) v4i*)(l);
  __builtin_amdgcn_global_load_async_to_lds_b128(gp, lp, 0, 0);
  __builtin_amdgcn_global_load_async_to_lds_b128(gp + 1, lp + 1, 0, 0);
#else
  const v8h* gpv = (const v8h*)g;
  *(v8h*)l       = gpv[0];
  *(v8h*)(l + 8) = gpv[1];
#endif
}

__device__ __forceinline__ void async_fence() {
#if USE_ASYNC_LDS
  asm volatile("s_wait_asynccnt 0x0" ::: "memory");
#endif
}

// ---------------------------------------------------------------------------
// 1) NCDHW -> (b,n,c) transpose + positional embedding (tiled, coalesced)
// ---------------------------------------------------------------------------
__global__ void k_transpose_pos(const float* __restrict__ x,
                                const float* __restrict__ pos,
                                float* __restrict__ xt) {
  __shared__ float tile[32][33];
  int b  = blockIdx.z;
  int p0 = blockIdx.x * 32;
  int c0 = blockIdx.y * 32;
  int tx = threadIdx.x, ty = threadIdx.y;
  for (int j = 0; j < 4; ++j) {
    int c = c0 + ty + j * 8;
    tile[ty + j * 8][tx] = x[((size_t)b * Cc + c) * Nn + p0 + tx];
  }
  __syncthreads();
  for (int j = 0; j < 4; ++j) {
    int p = p0 + ty + j * 8;
    int c = c0 + tx;
    xt[((size_t)b * Nn + p) * Cc + c] = tile[tx][ty + j * 8] + pos[(size_t)p * Cc + c];
  }
}

// ---------------------------------------------------------------------------
// 2) LayerNorm over C per token -> f16 activations
// ---------------------------------------------------------------------------
__global__ void k_layernorm(const float* __restrict__ xt,
                            const float* __restrict__ lnw,
                            const float* __restrict__ lnb,
                            _Float16* __restrict__ xnh) {
  __shared__ float red[256];
  size_t row = blockIdx.x;
  int t = threadIdx.x;
  float v = xt[row * Cc + t];
  red[t] = v; __syncthreads();
  for (int s = 128; s > 0; s >>= 1) { if (t < s) red[t] += red[t + s]; __syncthreads(); }
  float mu = red[0] * (1.0f / Cc);
  __syncthreads();
  float d = v - mu;
  red[t] = d * d; __syncthreads();
  for (int s = 128; s > 0; s >>= 1) { if (t < s) red[t] += red[t + s]; __syncthreads(); }
  float var = red[0] * (1.0f / Cc);
  xnh[row * Cc + t] = (_Float16)(d * rsqrtf(var + EPSv) * lnw[t] + lnb[t]);
}

// ---------------------------------------------------------------------------
// 3) Weight conversions f32 -> f16 (and conv rearrange (co,ci,27)->(tap,co,ci))
// ---------------------------------------------------------------------------
__global__ void k_cvt(const float* __restrict__ src, _Float16* __restrict__ dst, int n) {
  int i = blockIdx.x * 256 + threadIdx.x;
  if (i < n) dst[i] = (_Float16)src[i];
}

__global__ void k_cvt_conv(const float* __restrict__ src, _Float16* __restrict__ dst) {
  int i = blockIdx.x * 256 + threadIdx.x;     // total 256*256*27
  if (i >= Cc * Cc * 27) return;
  int co = i / (Cc * 27);
  int r  = i - co * (Cc * 27);
  int ci = r / 27;
  int tp = r - ci * 27;
  dst[((size_t)tp * Cc + co) * Cc + ci] = (_Float16)src[i];
}

// ---------------------------------------------------------------------------
// 4) Generic WMMA GEMM:  C[M,N] = A[M,K] * B[N,K]^T     (A,B f16 row-major)
//    Block 256 threads = 8 waves (4 in M x 2 in N). Block tile 128x128,
//    wave tile 32x64 (2 A-frags x 4 B-frags), K-step 32 staged in LDS.
//    MODE: 0 = f16 out (ld=N)
//          1 = f32 out into epa[:, coloff+gn] + bias
//          2 = bias + skip, scatter to NCDHW f32 out
// ---------------------------------------------------------------------------
template <int MODE>
__global__ void k_gemm_wmma(const _Float16* __restrict__ A,
                            const _Float16* __restrict__ Bw,
                            const float* __restrict__ bias,
                            const float* __restrict__ aux,
                            _Float16* __restrict__ outh,
                            float* __restrict__ outf,
                            int M, int N, int K, int coloff) {
  __shared__ __align__(16) _Float16 As[128][40];
  __shared__ __align__(16) _Float16 Bs[128][40];
  int m0 = blockIdx.x * 128;
  int n0 = blockIdx.y * 128;
  int t = threadIdx.x;
  int lane = t & 31, wv = t >> 5;
  int half = lane >> 4, lr = lane & 15;
  int rm0 = (wv & 3) * 32;        // wave row base within tile
  int cn0 = (wv >> 2) * 64;       // wave col base within tile
  v8f acc[2][4] = {};
  int ra = t >> 1, sga = (t & 1) * 16;

  for (int kb = 0; kb < K; kb += 32) {
    stage16(A  + (size_t)(m0 + ra) * K + kb + sga, &As[ra][sga]);
    stage16(Bw + (size_t)(n0 + ra) * K + kb + sga, &Bs[ra][sga]);
    if (kb + 32 < K)
      __builtin_prefetch(A + (size_t)(m0 + ra) * K + kb + 32 + sga, 0, 1);
    async_fence();
    __syncthreads();
    HFrag af[2];   // A 16x32: lanes 0-15 K{0..7,16..23}, lanes 16-31 K{8..15,24..31}
#pragma unroll
    for (int i = 0; i < 2; ++i) {
      af[i].h[0] = *(const v8h*)&As[rm0 + i * 16 + lr][half * 8];
      af[i].h[1] = *(const v8h*)&As[rm0 + i * 16 + lr][16 + half * 8];
    }
#pragma unroll
    for (int j = 0; j < 4; ++j) {
      HFrag bf;    // B 32x16: lanes 0-15 K{0..7,16..23} of col lr, lanes 16-31 K{8..15,24..31}
      bf.h[0] = *(const v8h*)&Bs[cn0 + j * 16 + lr][half * 8];
      bf.h[1] = *(const v8h*)&Bs[cn0 + j * 16 + lr][16 + half * 8];
#pragma unroll
      for (int i = 0; i < 2; ++i)
        acc[i][j] = __builtin_amdgcn_wmma_f32_16x16x32_f16(
            false, af[i].v, false, bf.v, (short)0, acc[i][j], false, false);
    }
    __syncthreads();
  }

#pragma unroll
  for (int i = 0; i < 2; ++i) {
#pragma unroll
    for (int j = 0; j < 4; ++j) {
#pragma unroll
      for (int vv = 0; vv < 8; ++vv) {
        int gm = m0 + rm0 + i * 16 + vv + half * 8;
        int gn = n0 + cn0 + j * 16 + lr;
        float val = acc[i][j][vv];
        if (MODE == 0) {
          outh[(size_t)gm * N + gn] = (_Float16)val;
        } else if (MODE == 1) {
          outf[(size_t)gm * Cc + coloff + gn] = val + bias[gn];
        } else {
          int bb = gm >> 12, p = gm & (Nn - 1);
          val += bias[gn] + aux[(size_t)gm * Cc + gn];
          outf[(((size_t)bb * Cc) + gn) * Nn + p] = val;
        }
      }
    }
  }
}

// ---------------------------------------------------------------------------
// 5) Implicit-GEMM 3x3x3 conv (27 shifted taps), channels-last f16 input.
//    MODE 1: BN + LeakyReLU ; MODE 2: BN + skip + LeakyReLU.
// ---------------------------------------------------------------------------
template <int MODE>
__global__ void k_conv_wmma(const _Float16* __restrict__ in,   // (b,n,c) f16
                            const _Float16* __restrict__ w,    // (27,co,ci) f16
                            const float* __restrict__ bnw, const float* __restrict__ bnb,
                            const float* __restrict__ bnrm, const float* __restrict__ bnrv,
                            const float* __restrict__ skip,
                            _Float16* __restrict__ outh) {
  __shared__ __align__(16) _Float16 As[128][40];
  __shared__ __align__(16) _Float16 Bs[128][40];
  int m0 = blockIdx.x * 128;
  int n0 = blockIdx.y * 128;
  int b  = blockIdx.z;
  int t = threadIdx.x;
  int lane = t & 31, wv = t >> 5;
  int half = lane >> 4, lr = lane & 15;
  int rm0 = (wv & 3) * 32;
  int cn0 = (wv >> 2) * 64;
  v8f acc[2][4] = {};
  int ra = t >> 1, sga = (t & 1) * 16;
  int pr = m0 + ra;
  int z0 = pr >> 8, y0 = (pr >> 4) & 15, x0 = pr & 15;

  for (int tap = 0; tap < 27; ++tap) {
    int dz = tap / 9 - 1, dy = (tap / 3) % 3 - 1, dx = tap % 3 - 1;
    int z = z0 + dz, y = y0 + dy, xx = x0 + dx;
    bool ok = ((unsigned)z < 16u) && ((unsigned)y < 16u) && ((unsigned)xx < 16u);
    size_t abase = ok ? (((size_t)b * Nn + ((z << 8) | (y << 4) | xx)) * Cc) : 0;
    for (int kb = 0; kb < Cc; kb += 32) {
      if (ok) {
        stage16(in + abase + kb + sga, &As[ra][sga]);
      } else {
        v8h zz = {};
        *(v8h*)&As[ra][sga]     = zz;
        *(v8h*)&As[ra][sga + 8] = zz;
      }
      stage16(w + (((size_t)tap * Cc) + n0 + ra) * Cc + kb + sga, &Bs[ra][sga]);
      async_fence();
      __syncthreads();
      HFrag af[2];
#pragma unroll
      for (int i = 0; i < 2; ++i) {
        af[i].h[0] = *(const v8h*)&As[rm0 + i * 16 + lr][half * 8];
        af[i].h[1] = *(const v8h*)&As[rm0 + i * 16 + lr][16 + half * 8];
      }
#pragma unroll
      for (int j = 0; j < 4; ++j) {
        HFrag bf;
        bf.h[0] = *(const v8h*)&Bs[cn0 + j * 16 + lr][half * 8];
        bf.h[1] = *(const v8h*)&Bs[cn0 + j * 16 + lr][16 + half * 8];
#pragma unroll
        for (int i = 0; i < 2; ++i)
          acc[i][j] = __builtin_amdgcn_wmma_f32_16x16x32_f16(
              false, af[i].v, false, bf.v, (short)0, acc[i][j], false, false);
      }
      __syncthreads();
    }
  }

#pragma unroll
  for (int j = 0; j < 4; ++j) {
    int gn = n0 + cn0 + j * 16 + lr;
    float scale = rsqrtf(bnrv[gn] + EPSv) * bnw[gn];
    float shift = bnb[gn] - bnrm[gn] * scale;
#pragma unroll
    for (int i = 0; i < 2; ++i) {
#pragma unroll
      for (int vv = 0; vv < 8; ++vv) {
        int p = m0 + rm0 + i * 16 + vv + half * 8;
        float val = acc[i][j][vv] * scale + shift;
        size_t oi = ((size_t)b * Nn + p) * Cc + gn;
        if (MODE == 2) val += skip[oi];
        val = val >= 0.f ? val : 0.01f * val;
        outh[oi] = (_Float16)val;
      }
    }
  }
}

// ---------------------------------------------------------------------------
// 6) q/k L2 norms over n -> reciprocal norms
// ---------------------------------------------------------------------------
__global__ void k_qknorm(const _Float16* __restrict__ qkvv,
                         float* __restrict__ rq, float* __restrict__ rk) {
  __shared__ float r1[256], r2[256];
  int id = blockIdx.x;                 // (b*8+h)*32+c
  int c = id & 31, h = (id >> 5) & 7, b = id >> 8;
  int t = threadIdx.x;
  float sq = 0.f, sk = 0.f;
  for (int p = t; p < Nn; p += 256) {
    const _Float16* base = qkvv + ((size_t)(b * Nn + p)) * 1024 + h * 32 + c;
    float qv = (float)base[0];
    float kv = (float)base[256];
    sq += qv * qv; sk += kv * kv;
  }
  r1[t] = sq; r2[t] = sk; __syncthreads();
  for (int s = 128; s > 0; s >>= 1) {
    if (t < s) { r1[t] += r1[t + s]; r2[t] += r2[t + s]; }
    __syncthreads();
  }
  if (t == 0) {
    rq[id] = 1.f / fmaxf(sqrtf(r1[0]), 1e-12f);
    rk[id] = 1.f / fmaxf(sqrtf(r2[0]), 1e-12f);
  }
}

// ---------------------------------------------------------------------------
// 7) Linformer projections kp = k.E^T + E_b, vp = vsa.F^T + F_b
// ---------------------------------------------------------------------------
__global__ void k_kpvp(const _Float16* __restrict__ qkvv,
                       const float* __restrict__ Ew, const float* __restrict__ Eb,
                       const float* __restrict__ Fw, const float* __restrict__ Fb,
                       float* __restrict__ kp, float* __restrict__ vp) {
  __shared__ float vals[256];
  __shared__ float red[256];
  int id = blockIdx.x;
  int sel = blockIdx.y;                // 0 -> kp (k,E); 1 -> vp (vsa,F)
  int c = id & 31, h = (id >> 5) & 7, b = id >> 8;
  int t = threadIdx.x;
  int p = t & 63, part = t >> 6;
  const float* W  = sel ? Fw : Ew;
  const float* Bi = sel ? Fb : Eb;
  float* out      = sel ? vp : kp;
  int off = sel ? 3 * Cc : 1 * Cc;
  float acc = 0.f;
  for (int nb = 0; nb < Nn; nb += 256) {
    vals[t] = (float)qkvv[((size_t)(b * Nn + nb + t)) * 1024 + off + h * 32 + c];
    __syncthreads();
    const float* wrow = W + (size_t)p * Nn + nb + part * 64;
    const float* vv = vals + part * 64;
#pragma unroll 8
    for (int i = 0; i < 64; ++i) acc += vv[i] * wrow[i];
    __syncthreads();
  }
  red[t] = acc; __syncthreads();
  if (part == 0)
    out[(size_t)id * 64 + p] = red[p] + red[p + 64] + red[p + 128] + red[p + 192] + Bi[p];
}

// ---------------------------------------------------------------------------
// 8) Channel attention 32x32 per (b,h): softmax(qn.kn^T * temp1)
// ---------------------------------------------------------------------------
__global__ void k_aca(const _Float16* __restrict__ qkvv,
                      const float* __restrict__ rq, const float* __restrict__ rk,
                      const float* __restrict__ temp1,
                      float* __restrict__ aca) {
  __shared__ _Float16 qs[32][256];
  __shared__ _Float16 ks[32][256];
  __shared__ float G[32][33];
  int bh = blockIdx.x;
  int h = bh & 7, b = bh >> 3;
  int t = threadIdx.x;
  int c = t >> 3, e0 = (t & 7) * 4;
  float accv[4] = {0.f, 0.f, 0.f, 0.f};
  for (int nb = 0; nb < Nn; nb += 256) {
    for (int idx = t; idx < 32 * 256; idx += 256) {
      int cc = idx >> 8, nn2 = idx & 255;
      const _Float16* bp = qkvv + ((size_t)(b * Nn + nb + nn2)) * 1024 + h * 32;
      qs[cc][nn2] = bp[cc];
      ks[cc][nn2] = bp[256 + cc];
    }
    __syncthreads();
#pragma unroll
    for (int j = 0; j < 4; ++j) {
      int e = e0 + j;
      float a = 0.f;
      for (int nn2 = 0; nn2 < 256; ++nn2)
        a += (float)qs[c][nn2] * (float)ks[e][nn2];
      accv[j] += a;
    }
    __syncthreads();
  }
  int idbase = bh * 32;
  float t1 = temp1[h];
  for (int j = 0; j < 4; ++j)
    G[c][e0 + j] = accv[j] * rq[idbase + c] * rk[idbase + e0 + j] * t1;
  __syncthreads();
  if (t < 32) {
    float m = -1e30f;
    for (int e = 0; e < 32; ++e) m = fmaxf(m, G[t][e]);
    float s = 0.f;
    for (int e = 0; e < 32; ++e) s += __expf(G[t][e] - m);
    float inv = 1.f / s;
    for (int e = 0; e < 32; ++e)
      aca[((size_t)idbase + t) * 32 + e] = __expf(G[t][e] - m) * inv;
  }
}

// ---------------------------------------------------------------------------
// 9) x_ca = a_ca @ vca, written channels-last f16
// ---------------------------------------------------------------------------
__global__ void k_xca(const _Float16* __restrict__ qkvv,
                      const float* __restrict__ aca,
                      _Float16* __restrict__ xca) {
  __shared__ float vs[256];
  size_t row = blockIdx.x;             // b*Nn + n
  int b = (int)(row >> 12);
  int t = threadIdx.x;
  int h = t >> 5, c = t & 31;
  vs[t] = (float)qkvv[row * 1024 + 2 * Cc + t];
  __syncthreads();
  const float* a = aca + ((size_t)(b * 8 + h) * 32 + c) * 32;
  float o = 0.f;
#pragma unroll 8
  for (int e = 0; e < 32; ++e) o += a[e] * vs[h * 32 + e];
  xca[row * Cc + t] = (_Float16)o;
}

// ---------------------------------------------------------------------------
// 10) Spatial attention, fused softmax: per (b,h,n): softmax(qn.kp*temp2).vp^T
// ---------------------------------------------------------------------------
__global__ void k_sa(const _Float16* __restrict__ qkvv,
                     const float* __restrict__ rq,
                     const float* __restrict__ kp, const float* __restrict__ vp,
                     const float* __restrict__ temp2,
                     _Float16* __restrict__ xsa) {
  __shared__ float kpl[32][64];
  __shared__ float vpl[32][64];
  __shared__ float qb[4][32];
  __shared__ float sb[4][64];
  __shared__ float smax[4], ssum[4];
  int bh = blockIdx.x;
  int h = bh & 7, b = bh >> 3;
  int nbase = blockIdx.y * 64;
  int t = threadIdx.x;
  for (int idx = t; idx < 2048; idx += 256) {
    int c = idx >> 6, p = idx & 63;
    kpl[c][p] = kp[((size_t)bh * 32 + c) * 64 + p];
    vpl[c][p] = vp[((size_t)bh * 32 + c) * 64 + p];
  }
  __syncthreads();
  float t2 = temp2[h];
  int nn = t >> 6, p = t & 63;
  for (int it = 0; it < 64; it += 4) {
    if (t < 128) {
      int nn2 = t >> 5, c = t & 31;
      int n2 = nbase + it + nn2;
      qb[nn2][c] = (float)qkvv[((size_t)(b * Nn + n2)) * 1024 + h * 32 + c]
                   * rq[(size_t)bh * 32 + c];
    }
    __syncthreads();
    float s = 0.f;
#pragma unroll 8
    for (int c = 0; c < 32; ++c) s += qb[nn][c] * kpl[c][p];
    s *= t2;
    sb[nn][p] = s;
    __syncthreads();
    if (t < 4) {
      float m = -1e30f;
      for (int pp = 0; pp < 64; ++pp) m = fmaxf(m, sb[t][pp]);
      float su = 0.f;
      for (int pp = 0; pp < 64; ++pp) su += __expf(sb[t][pp] - m);
      smax[t] = m; ssum[t] = su;
    }
    __syncthreads();
    float a = __expf(s - smax[nn]) / ssum[nn];
    sb[nn][p] = a;
    __syncthreads();
    if (t < 128) {
      int nn2 = t >> 5, c = t & 31;
      float o = 0.f;
#pragma unroll 8
      for (int pp = 0; pp < 64; ++pp) o += sb[nn2][pp] * vpl[c][pp];
      int n2 = nbase + it + nn2;
      xsa[((size_t)(b * Nn + n2)) * Cc + h * 32 + c] = (_Float16)o;
    }
    __syncthreads();
  }
}

// ---------------------------------------------------------------------------
// 11) skip = xt + gamma*epa  (f32 + f16 copies)
// ---------------------------------------------------------------------------
__global__ void k_skip(const float* __restrict__ xt, const float* __restrict__ epa,
                       const float* __restrict__ gamma,
                       float* __restrict__ skf, _Float16* __restrict__ skh) {
  size_t idx = (size_t)blockIdx.x * 256 + threadIdx.x;
  int c = threadIdx.x;                  // Cc == blockDim.x
  float s = xt[idx] + gamma[c] * epa[idx];
  skf[idx] = s;
  skh[idx] = (_Float16)s;
}

// ---------------------------------------------------------------------------
// Host orchestration
// ---------------------------------------------------------------------------
extern "C" void kernel_launch(void* const* d_in, const int* in_sizes, int n_in,
                              void* d_out, int out_size, void* d_ws, size_t ws_size,
                              hipStream_t stream) {
  (void)in_sizes; (void)n_in; (void)out_size; (void)ws_size;
  const float* x      = (const float*)d_in[0];
  const float* ln_w   = (const float*)d_in[1];
  const float* ln_b   = (const float*)d_in[2];
  const float* gamma  = (const float*)d_in[3];
  const float* pos    = (const float*)d_in[4];
  const float* qkvv_w = (const float*)d_in[5];
  const float* E_w    = (const float*)d_in[6];
  const float* E_b    = (const float*)d_in[7];
  const float* F_w    = (const float*)d_in[8];
  const float* F_b    = (const float*)d_in[9];
  const float* temp1  = (const float*)d_in[10];
  const float* temp2  = (const float*)d_in[11];
  const float* op1_w  = (const float*)d_in[12];
  const float* op1_b  = (const float*)d_in[13];
  const float* op2_w  = (const float*)d_in[14];
  const float* op2_b  = (const float*)d_in[15];
  const float* c1_w   = (const float*)d_in[16];
  const float* c2_w   = (const float*)d_in[17];
  const float* bn1_w  = (const float*)d_in[18];
  const float* bn1_b  = (const float*)d_in[19];
  const float* bn1_rm = (const float*)d_in[20];
  const float* bn1_rv = (const float*)d_in[21];
  const float* bn2_w  = (const float*)d_in[22];
  const float* bn2_b  = (const float*)d_in[23];
  const float* bn2_rm = (const float*)d_in[24];
  const float* bn2_rv = (const float*)d_in[25];
  const float* c8_w   = (const float*)d_in[26];
  const float* c8_b   = (const float*)d_in[27];
  float* out = (float*)d_out;

  char* ws = (char*)d_ws;
  size_t off = 0;
  auto take = [&](size_t bytes) -> char* {
    char* p = ws + off;
    off += (bytes + 255) & ~(size_t)255;
    return p;
  };
  const size_t BNC = (size_t)Bn * Nn * Cc;
  float*    xt    = (float*)take(BNC * 4);
  _Float16* xnh   = (_Float16*)take(BNC * 2);
  _Float16* qkvvh = (_Float16*)take(BNC * 4 * 2);       // (b,n,4C) f16
  _Float16* wqkvv = (_Float16*)take((size_t)4 * Cc * Cc * 2);
  _Float16* wop1  = (_Float16*)take((size_t)128 * Cc * 2);
  _Float16* wop2  = (_Float16*)take((size_t)128 * Cc * 2);
  _Float16* wc1   = (_Float16*)take((size_t)27 * Cc * Cc * 2);
  _Float16* wc2   = (_Float16*)take((size_t)27 * Cc * Cc * 2);
  _Float16* wc8   = (_Float16*)take((size_t)Cc * Cc * 2);
  float*    rq    = (float*)take(2048 * 4);
  float*    rk    = (float*)take(2048 * 4);
  float*    aca   = (float*)take((size_t)2048 * 32 * 4);
  float*    kp    = (float*)take((size_t)2048 * 64 * 4);
  float*    vp    = (float*)take((size_t)2048 * 64 * 4);
  _Float16* xca   = (_Float16*)take(BNC * 2);
  _Float16* xsa   = (_Float16*)take(BNC * 2);
  float*    epa   = (float*)take(BNC * 4);
  float*    skf   = (float*)take(BNC * 4);
  _Float16* skh   = (_Float16*)take(BNC * 2);
  _Float16* o1    = (_Float16*)take(BNC * 2);
  _Float16* o2    = (_Float16*)take(BNC * 2);

  // 1-2: transpose + pos embed, LayerNorm
  k_transpose_pos<<<dim3(Nn / 32, Cc / 32, Bn), dim3(32, 8), 0, stream>>>(x, pos, xt);
  k_layernorm<<<Bn * Nn, 256, 0, stream>>>(xt, ln_w, ln_b, xnh);

  // 3: weight conversions
  k_cvt<<<(4 * Cc * Cc + 255) / 256, 256, 0, stream>>>(qkvv_w, wqkvv, 4 * Cc * Cc);
  k_cvt<<<(128 * Cc + 255) / 256, 256, 0, stream>>>(op1_w, wop1, 128 * Cc);
  k_cvt<<<(128 * Cc + 255) / 256, 256, 0, stream>>>(op2_w, wop2, 128 * Cc);
  k_cvt<<<(Cc * Cc + 255) / 256, 256, 0, stream>>>(c8_w, wc8, Cc * Cc);
  k_cvt_conv<<<(Cc * Cc * 27 + 255) / 256, 256, 0, stream>>>(c1_w, wc1);
  k_cvt_conv<<<(Cc * Cc * 27 + 255) / 256, 256, 0, stream>>>(c2_w, wc2);

  // 4: QKVV projection (WMMA): (32768 x 1024 x 256)
  k_gemm_wmma<0><<<dim3((Bn * Nn) / 128, (4 * Cc) / 128), 256, 0, stream>>>(
      xnh, wqkvv, nullptr, nullptr, qkvvh, nullptr, Bn * Nn, 4 * Cc, Cc, 0);

  // 5: attention small ops
  k_qknorm<<<Bn * NHh * DHh, 256, 0, stream>>>(qkvvh, rq, rk);
  k_kpvp<<<dim3(Bn * NHh * DHh, 2), 256, 0, stream>>>(qkvvh, E_w, E_b, F_w, F_b, kp, vp);
  k_aca<<<Bn * NHh, 256, 0, stream>>>(qkvvh, rq, rk, temp1, aca);
  k_xca<<<Bn * Nn, 256, 0, stream>>>(qkvvh, aca, xca);
  k_sa<<<dim3(Bn * NHh, Nn / 64), 256, 0, stream>>>(qkvvh, rq, kp, vp, temp2, xsa);

  // 6: op1/op2 projections into epa (WMMA), N=128 -> one column tile
  k_gemm_wmma<1><<<dim3((Bn * Nn) / 128, 1), 256, 0, stream>>>(
      xsa, wop1, op1_b, nullptr, nullptr, epa, Bn * Nn, 128, Cc, 0);
  k_gemm_wmma<1><<<dim3((Bn * Nn) / 128, 1), 256, 0, stream>>>(
      xca, wop2, op2_b, nullptr, nullptr, epa, Bn * Nn, 128, Cc, 128);

  // 7: residual skip
  k_skip<<<Bn * Nn, 256, 0, stream>>>(xt, epa, gamma, skf, skh);

  // 8: UnetResBlock convs (implicit GEMM, WMMA)
  k_conv_wmma<1><<<dim3(Nn / 128, Cc / 128, Bn), 256, 0, stream>>>(
      skh, wc1, bn1_w, bn1_b, bn1_rm, bn1_rv, nullptr, o1);
  k_conv_wmma<2><<<dim3(Nn / 128, Cc / 128, Bn), 256, 0, stream>>>(
      o1, wc2, bn2_w, bn2_b, bn2_rm, bn2_rv, skf, o2);

  // 9: 1x1x1 conv + bias + skip -> NCDHW f32 output (WMMA)
  k_gemm_wmma<2><<<dim3((Bn * Nn) / 128, Cc / 128), 256, 0, stream>>>(
      o2, wc8, c8_b, skf, nullptr, out, Bn * Nn, Cc, Cc, 0);
}
